// MultiFactor_4123168604946
// MI455X (gfx1250) — matrix operand order repaired and checked by reading, not compiled, not run on hardware
//
#include <hip/hip_runtime.h>
#include <hip/hip_bf16.h>
#include <stdint.h>

// ---- problem constants (from reference) ----
#define B_ 8
#define T_ 768
#define H_ 128
#define F_ 10
#define TI_ (T_/16)   // 48 i/j tiles
#define KC_ (H_/32)   // 4 k-chunks over H
#define JC_ (T_/32)   // 24 j-chunks over T

typedef __attribute__((ext_vector_type(16))) __bf16          v16bf;
typedef __attribute__((ext_vector_type(8)))  float           v8f;
typedef __attribute__((ext_vector_type(8)))  unsigned short  us8;
typedef __attribute__((ext_vector_type(16))) unsigned short  us16;

__device__ __forceinline__ unsigned short f2bf(float x) {
  unsigned int u = __float_as_uint(x);
  u += 0x7FFFu + ((u >> 16) & 1u);   // round-to-nearest-even
  return (unsigned short)(u >> 16);
}

__device__ __forceinline__ v8f vzero8() {
  v8f z = {0.f,0.f,0.f,0.f,0.f,0.f,0.f,0.f};
  return z;
}

// ---- gfx1250 async global->LDS copy (one instr = 32 lanes x 16B = 512B) ----
// LDS address is LDS base plus the per-lane VDST VGPR value; LDS aperture keeps the wave-relative
// byte offset in the low 32 bits of a generic LDS pointer.
__device__ __forceinline__ void async_b128(unsigned int lds_byte_off,
                                           const void* gsrc) {
  asm volatile("global_load_async_to_lds_b128 %0, %1, off"
               :: "v"(lds_byte_off), "v"(gsrc)
               : "memory");
}
__device__ __forceinline__ void wait_async0() {
  asm volatile("s_wait_asynccnt 0x0" ::: "memory");
}
__device__ __forceinline__ unsigned int lds_off(const void* p) {
  return (unsigned int)(unsigned long long)p;
}

// A-fragment (16x32 bf16, row-major source, row stride ld elements).
// lane m=lane&15 holds row M=m; hi=lane>>4 selects 16 contiguous K
// {16*hi..16*hi+15} (matches load_b_frag).
__device__ __forceinline__ v16bf load_a_frag(const unsigned short* base,
                                             int ld, int lane) {
  const int m = lane & 15, hi = lane >> 4;
  const unsigned short* p = base + m * ld + hi * 16;
  us8 c0 = *(const us8*)(p);
  us8 c1 = *(const us8*)(p + 8);
  us16 v = __builtin_shufflevector(c0, c1, 0,1,2,3,4,5,6,7,8,9,10,11,12,13,14,15);
  return __builtin_bit_cast(v16bf, v);
}

// B-fragment (32x16 bf16). Source laid out [N][K] row-major with stride ld:
// lane holds column N=lane&15, 16 contiguous K starting at 16*(lane>>4).
__device__ __forceinline__ v16bf load_b_frag(const unsigned short* base,
                                             int ld, int lane) {
  const int n = lane & 15, hi = lane >> 4;
  us16 v = *(const us16*)(base + n * ld + hi * 16);
  return __builtin_bit_cast(v16bf, v);
}

__device__ __forceinline__ v8f wmma_bf16(v16bf a, v16bf b, v8f c) {
  return __builtin_amdgcn_wmma_f32_16x16x32_bf16(false, a, false, b,
                                                 (short)0, c, false, false);
}

// C/D 16x16 f32 layout: VGPR r, lane -> (M = r + 8*(lane>>4), N = lane&15)
__device__ __forceinline__ void store_c_f32(float* __restrict__ base, int ld,
                                            int lane, v8f acc) {
  const int n = lane & 15, hi = lane >> 4;
#pragma unroll
  for (int r = 0; r < 8; ++r) base[(r + 8 * hi) * ld + n] = acc[r];
}

__device__ __forceinline__ void store_c_bf16(unsigned short* __restrict__ base, int ld,
                                             int lane, v8f acc) {
  const int n = lane & 15, hi = lane >> 4;
#pragma unroll
  for (int r = 0; r < 8; ++r) base[(r + 8 * hi) * ld + n] = f2bf(acc[r]);
}

// ---------- kernel 0a: V fp32 -> bf16 (row-major) + bf16 transpose ----------
__global__ void k_prep_v(const float* __restrict__ V,
                         unsigned short* __restrict__ Vb,
                         unsigned short* __restrict__ VbT) {
  int idx = blockIdx.x * blockDim.x + threadIdx.x;
  if (idx >= B_ * T_ * H_) return;
  int h = idx % H_;
  int t = (idx / H_) % T_;
  int b = idx / (H_ * T_);
  unsigned short x = f2bf(V[idx]);
  Vb[idx] = x;
  VbT[(b * H_ + h) * T_ + t] = x;                    // [B][H][T]
}

// ---------- kernel 0b: W fp32 -> bf16 transposed: Wt[f][k'][h] = W[f][h][k'] ----------
__global__ void k_prep_w(const float* __restrict__ W,
                         unsigned short* __restrict__ Wt) {
  int idx = blockIdx.x * blockDim.x + threadIdx.x;
  if (idx >= F_ * H_ * H_) return;
  int k = idx % H_;
  int h = (idx / H_) % H_;
  int f = idx / (H_ * H_);
  Wt[(f * H_ + k) * H_ + h] = f2bf(W[idx]);
}

// ---------- kernel 1: U[b,f,i,:] = V[b,i,:] @ W_f ----------
// Block = 4 waves sharing (b,f): stage W_f^T (32 KB) once via async DMA.
__global__ void k_gemm_u(const unsigned short* __restrict__ Vb,
                         const unsigned short* __restrict__ Wt,
                         unsigned short* __restrict__ Ub) {
  __shared__ unsigned short sW[H_ * H_];            // 32 KB
  const int lane = threadIdx.x & 31;
  const int wave = threadIdx.x >> 5;
  int ig  = blockIdx.x % (TI_ / 4);
  int tmp = blockIdx.x / (TI_ / 4);
  int f = tmp % F_;
  int b = tmp / F_;
  int it = ig * 4 + wave;

  // stage W_f^T: 2048 x 16B chunks, 16 per thread
  {
    const unsigned short* g0 = Wt + (size_t)f * H_ * H_;
    unsigned int s0 = lds_off(sW);
#pragma unroll
    for (int q = 0; q < 16; ++q) {
      int c = threadIdx.x + q * 128;
      async_b128(s0 + (unsigned int)c * 16u, g0 + c * 8);
    }
    wait_async0();
    __syncthreads();
  }

  const unsigned short* Vrow = Vb + (size_t)(b * T_ + it * 16) * H_;

  v8f acc[8];
#pragma unroll
  for (int nt = 0; nt < 8; ++nt) acc[nt] = vzero8();

#pragma unroll
  for (int kc = 0; kc < KC_; ++kc) {
    v16bf a = load_a_frag(Vrow + kc * 32, H_, lane);
#pragma unroll
    for (int nt = 0; nt < 8; ++nt) {
      v16bf bfr = load_b_frag(sW + (nt * 16) * H_ + kc * 32, H_, lane);
      acc[nt] = wmma_bf16(a, bfr, acc[nt]);
    }
  }
  unsigned short* Urow = Ub + (size_t)((b * F_ + f) * T_ + it * 16) * H_;
#pragma unroll
  for (int nt = 0; nt < 8; ++nt) store_c_bf16(Urow + nt * 16, H_, lane, acc[nt]);
}

// ---------- kernel 2: scores[b,i,j] = max_f  U_f[i,:] . V[j,:] ----------
// Block = 4 waves sharing (b,i-tile): stage all 10 U_f i-tiles (40 KB) once.
__global__ void k_scores(const unsigned short* __restrict__ Ub,
                         const unsigned short* __restrict__ Vb,
                         float* __restrict__ S) {
  __shared__ unsigned short sU[F_ * 16 * H_];       // 40 KB
  const int lane = threadIdx.x & 31;
  const int wave = threadIdx.x >> 5;
  int jg  = blockIdx.x % (TI_ / 4);
  int tmp = blockIdx.x / (TI_ / 4);
  int it = tmp % TI_;
  int b  = tmp / TI_;
  int jt = jg * 4 + wave;

  // stage U tiles: each f-tile is 16x128 contiguous (4 KB); 2560 chunks, 20/thread
  {
    unsigned int s0 = lds_off(sU);
#pragma unroll
    for (int q = 0; q < 20; ++q) {
      int c = threadIdx.x + q * 128;
      int f = c >> 8;                                // 256 chunks per f
      int w = c & 255;
      const unsigned short* g =
          Ub + (size_t)((b * F_ + f) * T_ + it * 16) * H_ + w * 8;
      async_b128(s0 + (unsigned int)c * 16u, g);
    }
    wait_async0();
    __syncthreads();
  }

  // B-fragments: row-major V[j,:] is exactly the 128(H) x 16(j) B layout
  const unsigned short* Vj = Vb + (size_t)(b * T_ + jt * 16) * H_;
  v16bf bf[KC_];
#pragma unroll
  for (int kc = 0; kc < KC_; ++kc) bf[kc] = load_b_frag(Vj + kc * 32, H_, lane);

  v8f mx;
#pragma unroll
  for (int r = 0; r < 8; ++r) mx[r] = -3.402823466e38f;

  for (int f = 0; f < F_; ++f) {
    const unsigned short* Uf = sU + f * 16 * H_;
    v8f acc = vzero8();
#pragma unroll
    for (int kc = 0; kc < KC_; ++kc)
      acc = wmma_bf16(load_a_frag(Uf + kc * 32, H_, lane), bf[kc], acc);
#pragma unroll
    for (int r = 0; r < 8; ++r) mx[r] = fmaxf(mx[r], acc[r]);
  }
  float* Srow = S + (size_t)(b * T_ + it * 16) * T_ + jt * 16;
  store_c_f32(Srow, T_, lane, mx);
}

// ---------- kernel 3: row softmax (fp32) + bf16 convert; one wave per row ----------
__global__ void k_softmax(const float* __restrict__ S,
                          unsigned short* __restrict__ Ab) {
  int row  = (blockIdx.x * blockDim.x + threadIdx.x) >> 5;   // b*T + i
  int lane = threadIdx.x & 31;
  const float* s = S + (size_t)row * T_;

  float vals[T_ / 32];
  float m = -3.402823466e38f;
#pragma unroll
  for (int q = 0; q < T_ / 32; ++q) {
    vals[q] = s[lane + q * 32];
    m = fmaxf(m, vals[q]);
  }
#pragma unroll
  for (int off = 16; off > 0; off >>= 1) m = fmaxf(m, __shfl_xor(m, off, 32));

  float sum = 0.f;
#pragma unroll
  for (int q = 0; q < T_ / 32; ++q) {
    vals[q] = __expf(vals[q] - m);
    sum += vals[q];
  }
#pragma unroll
  for (int off = 16; off > 0; off >>= 1) sum += __shfl_xor(sum, off, 32);

  float inv = 1.0f / sum;
  unsigned short* a = Ab + (size_t)row * T_;
#pragma unroll
  for (int q = 0; q < T_ / 32; ++q) a[lane + q * 32] = f2bf(vals[q] * inv);
}

// ---------- kernel 4: out[b,i,h] = sum_j att[b,i,j] V[b,j,h] ----------
// Block = 4 waves sharing (b,i-tile): stage att i-tile row block (24 KB) once.
__global__ void k_out(const unsigned short* __restrict__ Ab,
                      const unsigned short* __restrict__ VbT,
                      float* __restrict__ Out) {
  __shared__ unsigned short sA[16 * T_];            // 24 KB
  const int lane = threadIdx.x & 31;
  const int wave = threadIdx.x >> 5;
  int hg  = blockIdx.x & 1;                         // two ht groups of 4
  int tmp = blockIdx.x >> 1;
  int it = tmp % TI_;
  int b  = tmp / TI_;
  int ht = hg * 4 + wave;

  // stage att rows i0..i0+15 (16x768 contiguous): 1536 chunks, 12/thread
  {
    const unsigned short* g0 = Ab + (size_t)(b * T_ + it * 16) * T_;
    unsigned int s0 = lds_off(sA);
#pragma unroll
    for (int q = 0; q < 12; ++q) {
      int c = threadIdx.x + q * 128;
      async_b128(s0 + (unsigned int)c * 16u, g0 + c * 8);
    }
    wait_async0();
    __syncthreads();
  }

  const unsigned short* Vt = VbT + (size_t)(b * H_ + ht * 16) * T_;  // [h][j]

  v8f acc = vzero8();
#pragma unroll 4
  for (int jc = 0; jc < JC_; ++jc) {
    v16bf a  = load_a_frag(sA + jc * 32, T_, lane);
    v16bf bb = load_b_frag(Vt + jc * 32, T_, lane);
    acc = wmma_bf16(a, bb, acc);
  }
  float* Orow = Out + (size_t)(b * T_ + it * 16) * H_ + ht * 16;
  store_c_f32(Orow, H_, lane, acc);
}

// ---------------------------------------------------------------------------
extern "C" void kernel_launch(void* const* d_in, const int* in_sizes, int n_in,
                              void* d_out, int out_size, void* d_ws, size_t ws_size,
                              hipStream_t stream) {
  const float* V = (const float*)d_in[0];   // [B,T,H] fp32
  const float* W = (const float*)d_in[1];   // [F,H,H] fp32
  float* Out = (float*)d_out;               // [B,T,H] fp32

  const size_t NV = (size_t)B_ * T_ * H_;      // 786432
  const size_t NW = (size_t)F_ * H_ * H_;      // 163840
  const size_t NU = (size_t)B_ * F_ * T_ * H_; // 7864320
  const size_t NS = (size_t)B_ * T_ * T_;      // 4718592

  // carve workspace (256B-aligned slices); total ~47.6 MB (L2-resident)
  uintptr_t p = ((uintptr_t)d_ws + 255) & ~(uintptr_t)255;
  auto take = [&](size_t bytes) {
    uintptr_t cur = p;
    p = (p + bytes + 255) & ~(uintptr_t)255;
    return (void*)cur;
  };
  unsigned short* Vb  = (unsigned short*)take(NV * 2);
  unsigned short* VbT = (unsigned short*)take(NV * 2);
  unsigned short* Wt  = (unsigned short*)take(NW * 2);
  unsigned short* Ub  = (unsigned short*)take(NU * 2);
  unsigned short* Ab  = (unsigned short*)take(NS * 2);
  float*          Sc  = (float*)take(NS * 4);
  (void)ws_size; (void)in_sizes; (void)n_in; (void)out_size;

  // 0) precision/layout prep
  k_prep_v<<<(NV + 255) / 256, 256, 0, stream>>>(V, Vb, VbT);
  k_prep_w<<<(NW + 255) / 256, 256, 0, stream>>>(W, Wt);

  // 1) U_f = V @ W_f   : 3840 waves, 4 waves/block share W_f via LDS
  k_gemm_u<<<(B_ * F_ * TI_) / 4, 128, 0, stream>>>(Vb, Wt, Ub);

  // 2) scores = max_f U_f V^T : 18432 waves, A-tiles staged in LDS
  k_scores<<<(B_ * TI_ * TI_) / 4, 128, 0, stream>>>(Ub, Vb, Sc);

  // 3) softmax rows : 6144 waves
  k_softmax<<<(B_ * T_) / 4, 128, 0, stream>>>(Sc, Ab);

  // 4) out = att @ V : 3072 waves, att tile staged in LDS
  k_out<<<(B_ * TI_ * (H_ / 16)) / 4, 128, 0, stream>>>(Ab, VbT, Out);
}
